// THHGNNLayer_37821482008813
// MI455X (gfx1250) — compile-verified, matrix operand order checked by reading, not run on hardware
//
#include <hip/hip_runtime.h>
#include <hip/hip_bf16.h>
#include <math.h>

#define FD 128            // feature dim
#define NN 100000         // nodes
#define EINC 300000       // incidence pairs per type
#define NHE 50000         // hyperedges per type

typedef __attribute__((ext_vector_type(16))) __bf16 v16bf;
typedef __attribute__((ext_vector_type(8)))  float  v8f;

union frag16 { v16bf v; uint4 q[2]; };

__device__ __forceinline__ __bf16 f2bf(float f) {
    union { float f; unsigned u; } a; a.f = f;
    unsigned u = a.u;
    u += 0x7FFFu + ((u >> 16) & 1u);           // round-to-nearest-even
    union { unsigned short s; __bf16 b; } r;
    r.s = (unsigned short)(u >> 16);
    return r.b;
}

__device__ __forceinline__ float wave_reduce_sum(float v) {
#pragma unroll
    for (int off = 16; off > 0; off >>= 1) v += __shfl_xor(v, off, 32);
    return v;
}

// ---------------------------------------------------------------------------
// GEMM: C[m, o] = cscale[m] * sum_k A[m,k] * W[o,k]     (A, W already bf16)
// One wave -> 16 rows x 128 cols: 8 f32 accumulators, 4 bf16 WMMAs each.
// C stored as f32 (Cf) or bf16 (Cb). Optional per-row type filter on stores.
// ---------------------------------------------------------------------------
__global__ __launch_bounds__(128)
void gemm_bf16_wmma(const __bf16* __restrict__ A, const __bf16* __restrict__ W,
                    float* __restrict__ Cf, __bf16* __restrict__ Cb,
                    int m_tiles,
                    const float* __restrict__ c_scale,
                    const int* __restrict__ row_filter, int filter_val) {
    int wave = threadIdx.x >> 5;
    int tile = blockIdx.x * 4 + wave;
    if (tile >= m_tiles) return;                 // whole-wave exit, EXEC stays full
    int lane = threadIdx.x & 31;
    int r    = lane & 15;
    int half = lane >> 4;

    // ---- A fragments: two contiguous 8-elem (16B) K-chunks per ISA layout ----
    const __bf16* Arow = A + (size_t)(tile * 16 + r) * FD;
    frag16 afrag[4];
#pragma unroll
    for (int kb = 0; kb < 4; ++kb) {
        int k0 = kb * 32 + half * 8;
        afrag[kb].q[0] = *(const uint4*)(Arow + k0);
        afrag[kb].q[1] = *(const uint4*)(Arow + k0 + 16);
    }

    // ---- accumulate over 8 output-column tiles ----
    v8f acc[8];
#pragma unroll
    for (int ob = 0; ob < 8; ++ob) {
#pragma unroll
        for (int v = 0; v < 8; ++v) acc[ob][v] = 0.0f;
        const __bf16* Wrow = W + (size_t)(ob * 16 + r) * FD;  // lane = B column r
#pragma unroll
        for (int kb = 0; kb < 4; ++kb) {
            frag16 bfrag;
            int k0 = kb * 32 + half * 8;
            bfrag.q[0] = *(const uint4*)(Wrow + k0);
            bfrag.q[1] = *(const uint4*)(Wrow + k0 + 16);
            acc[ob] = __builtin_amdgcn_wmma_f32_16x16x32_bf16(
                false, afrag[kb].v, false, bfrag.v, (short)0, acc[ob], false, false);
        }
    }

    // ---- store: acc[ob][v] -> C[tile*16 + half*8 + v][ob*16 + r] ----
    int mbase = tile * 16 + half * 8;
    float cs[8];
    bool  keep[8];
#pragma unroll
    for (int v = 0; v < 8; ++v) {
        int m = mbase + v;
        cs[v]   = c_scale ? c_scale[m] : 1.0f;
        keep[v] = row_filter ? (row_filter[m] == filter_val) : true;
    }
#pragma unroll
    for (int ob = 0; ob < 8; ++ob) {
#pragma unroll
        for (int v = 0; v < 8; ++v) {
            size_t idx = (size_t)(mbase + v) * FD + ob * 16 + r;
            if (keep[v]) {
                if (Cb) Cb[idx] = f2bf(acc[ob][v] * cs[v]);
                else    Cf[idx] = acc[ob][v] * cs[v];
            }
        }
    }
}

// ---------------------------------------------------------------------------
__global__ void fill_kernel(float* __restrict__ p, long long n, float v) {
    long long i = (long long)blockIdx.x * blockDim.x + threadIdx.x;
    long long s = (long long)gridDim.x * blockDim.x;
    for (; i < n; i += s) p[i] = v;
}

__global__ void cvt_bf16_kernel(const float* __restrict__ src,
                                __bf16* __restrict__ dst, long long n) {
    long long i = (long long)blockIdx.x * blockDim.x + threadIdx.x;
    long long s = (long long)gridDim.x * blockDim.x;
    for (; i < n; i += s) dst[i] = f2bf(src[i]);
}

// esum_bf[e,d] = bf16( esum[e,d] / max(cnt[e], 1) )
__global__ void scale_cvt_kernel(const float* __restrict__ src,
                                 const float* __restrict__ cnt,
                                 __bf16* __restrict__ dst, long long n) {
    long long i = (long long)blockIdx.x * blockDim.x + threadIdx.x;
    long long s = (long long)gridDim.x * blockDim.x;
    for (; i < n; i += s) {
        float inv = 1.0f / fmaxf(cnt[i >> 7], 1.0f);
        dst[i] = f2bf(src[i] * inv);
    }
}

// wave per incidence: esum[eid[e]] += h[nid[e]], ecnt[eid[e]] += 1
__global__ void scatter_edge_sum(const float* __restrict__ h,
                                 const int* __restrict__ nid,
                                 const int* __restrict__ eid,
                                 float* __restrict__ esum,
                                 float* __restrict__ ecnt, int n_inc) {
    int e = blockIdx.x * (blockDim.x >> 5) + (threadIdx.x >> 5);
    if (e >= n_inc) return;
    int lane = threadIdx.x & 31;
    int src = nid[e], dst = eid[e];
    const float4 v = ((const float4*)(h + (size_t)src * FD))[lane];
    float* ed = esum + (size_t)dst * FD + lane * 4;
    atomicAdd(ed + 0, v.x); atomicAdd(ed + 1, v.y);
    atomicAdd(ed + 2, v.z); atomicAdd(ed + 3, v.w);
    if (lane == 0) atomicAdd(ecnt + dst, 1.0f);
}

// wave per incidence: score[e] = (q[nid[e]] . k[eid[e]]) / sqrt(D)
__global__ void score_kernel(const float* __restrict__ q, const float* __restrict__ k,
                             const int* __restrict__ nid, const int* __restrict__ eid,
                             float* __restrict__ score, int n_inc) {
    int e = blockIdx.x * (blockDim.x >> 5) + (threadIdx.x >> 5);
    if (e >= n_inc) return;
    int lane = threadIdx.x & 31;
    float4 qa = ((const float4*)(q + (size_t)nid[e] * FD))[lane];
    float4 ka = ((const float4*)(k + (size_t)eid[e] * FD))[lane];
    float s = qa.x * ka.x + qa.y * ka.y + qa.z * ka.z + qa.w * ka.w;
    s = wave_reduce_sum(s);
    if (lane == 0) score[e] = s * 0.08838834764831845f; // 1/sqrt(128)
}

__device__ __forceinline__ void atomicMaxFloat(float* addr, float value) {
    if (value >= 0.0f) atomicMax((int*)addr, __float_as_int(value));
    else               atomicMin((unsigned int*)addr, __float_as_uint(value));
}

__global__ void seg_max_kernel(const float* __restrict__ score,
                               const int* __restrict__ nid,
                               float* __restrict__ nmax, int n_inc) {
    int e = blockIdx.x * blockDim.x + threadIdx.x;
    if (e < n_inc) atomicMaxFloat(&nmax[nid[e]], score[e]);
}

__global__ void exp_denom_kernel(float* __restrict__ score,
                                 const float* __restrict__ nmax,
                                 const int* __restrict__ nid,
                                 float* __restrict__ denom, int n_inc) {
    int e = blockIdx.x * blockDim.x + threadIdx.x;
    if (e >= n_inc) return;
    int n = nid[e];
    float ex = expf(score[e] - nmax[n]);
    score[e] = ex;
    atomicAdd(&denom[n], ex);
}

// wave per incidence: upd[nid[e]] += attn * v[eid[e]]
__global__ void scatter_msg(const float* __restrict__ score,
                            const float* __restrict__ denom,
                            const float* __restrict__ vmat,
                            const int* __restrict__ nid,
                            const int* __restrict__ eid,
                            float* __restrict__ upd, int n_inc) {
    int e = blockIdx.x * (blockDim.x >> 5) + (threadIdx.x >> 5);
    if (e >= n_inc) return;
    int lane = threadIdx.x & 31;
    int n = nid[e];
    float attn = score[e] / (denom[n] + 1e-12f);
    float4 v = ((const float4*)(vmat + (size_t)eid[e] * FD))[lane];
    float* d = upd + (size_t)n * FD + lane * 4;
    atomicAdd(d + 0, attn * v.x); atomicAdd(d + 1, attn * v.y);
    atomicAdd(d + 2, attn * v.z); atomicAdd(d + 3, attn * v.w);
}

// wave per node: out = LN(x + upd/4) * gamma + beta
__global__ void residual_ln_kernel(const float* __restrict__ x,
                                   const float* __restrict__ upd,
                                   const float* __restrict__ gamma,
                                   const float* __restrict__ beta,
                                   float* __restrict__ out, int n_nodes) {
    int n = blockIdx.x * (blockDim.x >> 5) + (threadIdx.x >> 5);
    if (n >= n_nodes) return;
    int lane = threadIdx.x & 31;
    float4 xa = ((const float4*)(x   + (size_t)n * FD))[lane];
    float4 ua = ((const float4*)(upd + (size_t)n * FD))[lane];
    float4 h = make_float4(xa.x + 0.25f * ua.x, xa.y + 0.25f * ua.y,
                           xa.z + 0.25f * ua.z, xa.w + 0.25f * ua.w);
    float mean = wave_reduce_sum(h.x + h.y + h.z + h.w) * (1.0f / FD);
    float4 d = make_float4(h.x - mean, h.y - mean, h.z - mean, h.w - mean);
    float var = wave_reduce_sum(d.x * d.x + d.y * d.y + d.z * d.z + d.w * d.w) * (1.0f / FD);
    float rstd = rsqrtf(var + 1e-5f);
    float4 g = ((const float4*)gamma)[lane];
    float4 b = ((const float4*)beta)[lane];
    float4 o = make_float4(d.x * rstd * g.x + b.x, d.y * rstd * g.y + b.y,
                           d.z * rstd * g.z + b.z, d.w * rstd * g.w + b.w);
    ((float4*)(out + (size_t)n * FD))[lane] = o;
}

// ---------------------------------------------------------------------------
static inline void launch_gemm(const __bf16* A, const __bf16* W,
                               float* Cf, __bf16* Cb, int M,
                               const float* cscale, const int* filt, int fv,
                               hipStream_t s) {
    int tiles = M / 16;
    gemm_bf16_wmma<<<(tiles + 3) / 4, 128, 0, s>>>(A, W, Cf, Cb, tiles, cscale, filt, fv);
}

extern "C" void kernel_launch(void* const* d_in, const int* in_sizes, int n_in,
                              void* d_out, int out_size, void* d_ws, size_t ws_size,
                              hipStream_t stream) {
    const float* x        = (const float*)d_in[0];
    const int*   type_ids = (const int*)d_in[1];
    const int*   nid[4] = { (const int*)d_in[2],  (const int*)d_in[5],
                            (const int*)d_in[8],  (const int*)d_in[11] };
    const int*   eid[4] = { (const int*)d_in[3],  (const int*)d_in[6],
                            (const int*)d_in[9],  (const int*)d_in[12] };
    const float* ew[4]  = { (const float*)d_in[4],  (const float*)d_in[7],
                            (const float*)d_in[10], (const float*)d_in[13] };
    const float* W_node = (const float*)d_in[14];
    const float* W_edge = (const float*)d_in[15];
    const float* Wq     = (const float*)d_in[16];
    const float* Wk     = (const float*)d_in[17];
    const float* Wv     = (const float*)d_in[18];
    const float* gamma  = (const float*)d_in[19];
    const float* beta   = (const float*)d_in[20];
    float* out = (float*)d_out;

    const size_t ND = (size_t)NN * FD;    // 12.8M
    const size_t HD = (size_t)NHE * FD;   // 6.4M
    const size_t DD = (size_t)FD * FD;    // 16384

    // f32 workspace
    float* ws     = (float*)d_ws;
    float* h_node = ws;                   // ND
    float* q_node = h_node + ND;          // ND
    float* upd    = q_node + ND;          // ND
    float* esum   = upd + ND;             // HD  (reused as k_e after esum_bf built)
    float* v_e    = esum + HD;            // HD
    float* ecnt   = v_e + HD;             // NHE
    float* score  = ecnt + NHE;           // EINC
    float* nmax   = score + EINC;         // NN
    float* ndenom = nmax + NN;            // NN
    float* k_e    = esum;

    // bf16 workspace (after f32 region)
    __bf16* bfws    = (__bf16*)(ndenom + NN);
    __bf16* x_bf    = bfws;               // ND
    __bf16* esum_bf = x_bf + ND;          // HD
    __bf16* ee2_bf  = esum_bf + HD;       // HD
    __bf16* wb      = ee2_bf + HD;        // 19*DD
    __bf16* Wn_bf   = wb;                 // 3*DD
    __bf16* We_bf   = Wn_bf + 3 * DD;     // 4*DD
    __bf16* Wq_bf   = We_bf + 4 * DD;     // 4*DD
    __bf16* Wk_bf   = Wq_bf + 4 * DD;     // 4*DD
    __bf16* Wv_bf   = Wk_bf + 4 * DD;     // 4*DD

    // one-time conversions (amortized: each element converted exactly once)
    cvt_bf16_kernel<<<4096, 256, 0, stream>>>(x, x_bf, (long long)ND);
    cvt_bf16_kernel<<<48, 256, 0, stream>>>(W_node, Wn_bf, (long long)(3 * DD));
    cvt_bf16_kernel<<<64, 256, 0, stream>>>(W_edge, We_bf, (long long)(4 * DD));
    cvt_bf16_kernel<<<64, 256, 0, stream>>>(Wq, Wq_bf, (long long)(4 * DD));
    cvt_bf16_kernel<<<64, 256, 0, stream>>>(Wk, Wk_bf, (long long)(4 * DD));
    cvt_bf16_kernel<<<64, 256, 0, stream>>>(Wv, Wv_bf, (long long)(4 * DD));

    // node-type projection: h_node[m] = x[m] @ W_node[type[m]]^T (filtered stores)
    for (int t = 0; t < 3; ++t)
        launch_gemm(x_bf, Wn_bf + t * DD, h_node, nullptr, NN, nullptr, type_ids, t, stream);

    fill_kernel<<<4096, 256, 0, stream>>>(upd, (long long)ND, 0.0f);

    const int inc_waves_grid = (EINC + 7) / 8;        // 8 waves / 256-thread block
    const int inc_thr_grid   = (EINC + 255) / 256;

    for (int t = 0; t < 4; ++t) {
        // Q projection for all nodes
        launch_gemm(x_bf, Wq_bf + t * DD, q_node, nullptr, NN, nullptr, nullptr, 0, stream);

        // hyperedge mean embedding (atomic accumulate, then scale+cvt to bf16)
        fill_kernel<<<2048, 256, 0, stream>>>(esum, (long long)HD, 0.0f);
        fill_kernel<<<256, 256, 0, stream>>>(ecnt, (long long)NHE, 0.0f);
        scatter_edge_sum<<<inc_waves_grid, 256, 0, stream>>>(h_node, nid[t], eid[t],
                                                             esum, ecnt, EINC);
        scale_cvt_kernel<<<2048, 256, 0, stream>>>(esum, ecnt, esum_bf, (long long)HD);

        // ee2 = (mean_emb @ W_edge^T) * edge_w  (stored bf16: GEMM-input only)
        launch_gemm(esum_bf, We_bf + t * DD, nullptr, ee2_bf, NHE, ew[t], nullptr, 0, stream);
        launch_gemm(ee2_bf, Wk_bf + t * DD, k_e, nullptr, NHE, nullptr, nullptr, 0, stream);
        launch_gemm(ee2_bf, Wv_bf + t * DD, v_e, nullptr, NHE, nullptr, nullptr, 0, stream);

        // attention scores + segment softmax + message scatter
        score_kernel<<<inc_waves_grid, 256, 0, stream>>>(q_node, k_e, nid[t], eid[t],
                                                         score, EINC);
        fill_kernel<<<256, 256, 0, stream>>>(nmax, (long long)NN, -3.402823466e38f);
        seg_max_kernel<<<inc_thr_grid, 256, 0, stream>>>(score, nid[t], nmax, EINC);
        fill_kernel<<<256, 256, 0, stream>>>(ndenom, (long long)NN, 0.0f);
        exp_denom_kernel<<<inc_thr_grid, 256, 0, stream>>>(score, nmax, nid[t],
                                                           ndenom, EINC);
        scatter_msg<<<inc_waves_grid, 256, 0, stream>>>(score, ndenom, v_e,
                                                        nid[t], eid[t], upd, EINC);
    }

    residual_ln_kernel<<<(NN + 7) / 8, 256, 0, stream>>>(x, upd, gamma, beta, out, NN);
}
